// CKAN_57870389346504
// MI455X (gfx1250) — matrix operand order repaired and checked by reading, not compiled, not run on hardware
//
#include <hip/hip_runtime.h>

#define DIM 64
#define TT  64

typedef __attribute__((ext_vector_type(16))) __bf16        v16bf;
typedef __attribute__((ext_vector_type(8)))  float         v8f;
typedef __attribute__((ext_vector_type(8)))  unsigned int  v8u;

__device__ __forceinline__ unsigned short f2bf(float x) {
    unsigned int u = __builtin_bit_cast(unsigned int, x);
    u += 0x7FFFu + ((u >> 16) & 1u);              // round-to-nearest-even
    return (unsigned short)(u >> 16);
}
__device__ __forceinline__ float bf2f(unsigned short h) {
    unsigned int u = ((unsigned int)h) << 16;
    return __builtin_bit_cast(float, u);
}
__device__ __forceinline__ unsigned int pk2(float a, float b) {
    return (unsigned int)f2bf(a) | ((unsigned int)f2bf(b) << 16);
}

// ---------------------------------------------------------------------------
// Kernel 0: pack W1^T / W2^T (fp32) into bf16 WMMA B-fragment layout.
// B (32x16, 16-bit), contiguous K order used here: lane L holds col N=L%16; elements e=0..15 -> K=e+(L>=16?16:0).
// Fragment index = ((kk*4 + n)*32 + lane)*8 + j, uint j packs K=2j,2j+1.
// ---------------------------------------------------------------------------
__global__ void ckan_pack_weights(const float* __restrict__ W1,   // [64][128]
                                  const float* __restrict__ W2,   // [64][64]
                                  unsigned int* __restrict__ gW1B,
                                  unsigned int* __restrict__ gW2B) {
    int tid = blockIdx.x * blockDim.x + threadIdx.x;
    int nthr = gridDim.x * blockDim.x;
    for (int idx = tid; idx < 4096; idx += nthr) {      // 4 kk * 4 n * 32 * 8
        int kk = idx >> 10, n = (idx >> 8) & 3, lane = (idx >> 3) & 31, j = idx & 7;
        int K = 2 * j + ((lane >= 16) ? 16 : 0);
        int k = 32 * kk + K;
        int c = n * 16 + (lane & 15);
        gW1B[idx] = pk2(W1[c * 128 + k], W1[c * 128 + k + 1]);   // W1T[k][c] = W1[c][k]
    }
    for (int idx = tid; idx < 2048; idx += nthr) {      // 2 kk * 4 n * 32 * 8
        int kk = idx >> 10, n = (idx >> 8) & 3, lane = (idx >> 3) & 31, j = idx & 7;
        int K = 2 * j + ((lane >= 16) ? 16 : 0);
        int k = 32 * kk + K;
        int c = n * 16 + (lane & 15);
        gW2B[idx] = pk2(W2[c * 64 + k], W2[c * 64 + k + 1]);
    }
}

// ---------------------------------------------------------------------------
// Kernel 1: one block per (b, call). 4 waves, each owning a 16-token M-tile.
// MLP via v_wmma_f32_16x16x32_bf16, then sigmoid -> softmax(T) -> weighted
// gather of tail embeddings. Result -> eatt[call][b][64].
// ---------------------------------------------------------------------------
__global__ __launch_bounds__(128) void ckan_attention(
    const float* __restrict__ emb, const float* __restrict__ rel,
    const float* __restrict__ W3,
    const unsigned int* __restrict__ gW1B, const unsigned int* __restrict__ gW2B,
    const int* __restrict__ user_h, const int* __restrict__ user_r, const int* __restrict__ user_t,
    const int* __restrict__ item_h, const int* __restrict__ item_r, const int* __restrict__ item_t,
    float* __restrict__ eatt, int Btot)
{
    __shared__ unsigned int  sW1B[4096];          // 16 KB packed B-frags (layer 1)
    __shared__ unsigned int  sW2B[2048];          //  8 KB packed B-frags (layer 2)
    __shared__ float         sW3[64];
    __shared__ unsigned short sX[64 * 136];       // [tok][128 bf16] (+8 pad)
    __shared__ unsigned short sY[64 * 72];        // layer-1 out bf16 (+8 pad)
    __shared__ unsigned short sZ[64 * 72];        // layer-2 out bf16 (+8 pad)
    __shared__ int   sH[64], sR[64], sT[64];
    __shared__ float sA[64], sWt[64], sPart[128];

    const int b    = blockIdx.x;
    const int call = blockIdx.y;                  // 0,1: user L0/L1 ; 2,3: item L0/L1
    const int tid  = threadIdx.x;
    const int lane = tid & 31;
    const int wv   = tid >> 5;

    const int* hp = (call < 2 ? user_h : item_h) + (size_t)(call & 1) * Btot * TT;
    const int* rp = (call < 2 ? user_r : item_r) + (size_t)(call & 1) * Btot * TT;
    const int* tp = (call < 2 ? user_t : item_t) + (size_t)(call & 1) * Btot * TT;

    // ---- stage packed weights + indices (all L2-resident) ----
    for (int i = tid; i < 4096; i += 128) sW1B[i] = gW1B[i];
    for (int i = tid; i < 2048; i += 128) sW2B[i] = gW2B[i];
    if (tid < 64) {
        sW3[tid] = W3[tid];
        sH[tid] = hp[(size_t)b * TT + tid];
        sR[tid] = rp[(size_t)b * TT + tid];
        sT[tid] = tp[(size_t)b * TT + tid];
    }
    __syncthreads();

    // ---- gather X = [h | r] -> bf16 rows in LDS (float4 coalesced loads) ----
    for (int piece = tid; piece < 64 * 32; piece += 128) {
        int tok = piece >> 5, p = piece & 31;
        float4 v;
        int col;
        if (p < 16) { v = ((const float4*)(emb + (size_t)sH[tok] * DIM))[p];      col = p * 4; }
        else        { v = ((const float4*)(rel + (size_t)sR[tok] * DIM))[p - 16]; col = 64 + (p - 16) * 4; }
        unsigned int* xr = (unsigned int*)(sX + tok * 136);
        xr[(col >> 1)]     = pk2(v.x, v.y);
        xr[(col >> 1) + 1] = pk2(v.z, v.w);
    }
    __syncthreads();

    const int tokBase = wv * 16;
    const int m  = lane & 15;
    const int lh = (lane >> 4) << 3;              // 0 / 8 : lane-half C row offset (K offset is 2*lh)

    // ---- layer 1: relu(X[16x128] @ W1T[128x64]) via 4x4 WMMA ----
    v16bf a1[4];
    {
        const unsigned int* rowp = (const unsigned int*)(sX + (tokBase + m) * 136);
        #pragma unroll
        for (int kk = 0; kk < 4; ++kk) {
            v8u u;
            #pragma unroll
            for (int j = 0; j < 8; ++j) {
                int e = 2 * j;
                int K = e + 2 * lh;                    // contiguous K = 16*(lane>=16) + e, matches B pack
                u[j] = rowp[(32 * kk + K) >> 1];
            }
            a1[kk] = __builtin_bit_cast(v16bf, u);
        }
    }
    #pragma unroll
    for (int n = 0; n < 4; ++n) {
        v8f acc = {0.f, 0.f, 0.f, 0.f, 0.f, 0.f, 0.f, 0.f};
        #pragma unroll
        for (int kk = 0; kk < 4; ++kk) {
            const v8u* bp = (const v8u*)&sW1B[((kk * 4 + n) * 32 + lane) * 8];
            v16bf bb = __builtin_bit_cast(v16bf, *bp);
            acc = __builtin_amdgcn_wmma_f32_16x16x32_bf16(false, a1[kk], false, bb,
                                                          (short)0, acc, false, false);
        }
        #pragma unroll
        for (int v = 0; v < 8; ++v) {                 // C layout: row = v + 8*(lane>=16)
            int tok = tokBase + v + lh;
            sY[tok * 72 + n * 16 + m] = f2bf(fmaxf(acc[v], 0.f));
        }
    }
    asm volatile("s_wait_dscnt 0" ::: "memory");      // Y stores visible before A-frag reads

    // ---- layer 2: relu(Y[16x64] @ W2T[64x64]) via 2x4 WMMA ----
    v16bf a2[2];
    {
        const unsigned int* rowp = (const unsigned int*)(sY + (tokBase + m) * 72);
        #pragma unroll
        for (int kk = 0; kk < 2; ++kk) {
            v8u u;
            #pragma unroll
            for (int j = 0; j < 8; ++j) {
                int e = 2 * j;
                int K = e + 2 * lh;
                u[j] = rowp[(32 * kk + K) >> 1];
            }
            a2[kk] = __builtin_bit_cast(v16bf, u);
        }
    }
    #pragma unroll
    for (int n = 0; n < 4; ++n) {
        v8f acc = {0.f, 0.f, 0.f, 0.f, 0.f, 0.f, 0.f, 0.f};
        #pragma unroll
        for (int kk = 0; kk < 2; ++kk) {
            const v8u* bp = (const v8u*)&sW2B[((kk * 4 + n) * 32 + lane) * 8];
            v16bf bb = __builtin_bit_cast(v16bf, *bp);
            acc = __builtin_amdgcn_wmma_f32_16x16x32_bf16(false, a2[kk], false, bb,
                                                          (short)0, acc, false, false);
        }
        #pragma unroll
        for (int v = 0; v < 8; ++v) {
            int tok = tokBase + v + lh;
            sZ[tok * 72 + n * 16 + m] = f2bf(fmaxf(acc[v], 0.f));
        }
    }
    __syncthreads();

    // ---- a = sigmoid(Z @ W3^T) ----
    if (tid < 64) {
        float s = 0.f;
        #pragma unroll 8
        for (int d = 0; d < 64; ++d) s += bf2f(sZ[tid * 72 + d]) * sW3[d];
        sA[tid] = 1.f / (1.f + __expf(-s));
    }
    __syncthreads();

    // ---- softmax over T=64 ----
    if (tid < 64) {
        float mx = -1e30f;
        for (int j = 0; j < 64; ++j) mx = fmaxf(mx, sA[j]);
        float sum = 0.f;
        for (int j = 0; j < 64; ++j) sum += __expf(sA[j] - mx);
        sWt[tid] = __expf(sA[tid] - mx) / sum;
    }
    __syncthreads();

    // ---- e[d] = sum_t w[t] * emb[t_idx[t]][d]  (coalesced rows, split over 2 halves) ----
    {
        int th = tid >> 6, d = tid & 63;
        float acc = 0.f;
        for (int t = th * 32; t < th * 32 + 32; ++t)
            acc += sWt[t] * emb[(size_t)sT[t] * DIM + d];
        sPart[th * 64 + d] = acc;
    }
    __syncthreads();
    if (tid < 64)
        eatt[((size_t)call * Btot + b) * DIM + tid] = sPart[tid] + sPart[64 + tid];
}

// ---------------------------------------------------------------------------
// Kernel 2: fuse head means + attention terms, sigmoid(dot(e_u, e_v)).
// ---------------------------------------------------------------------------
__global__ __launch_bounds__(64) void ckan_finalize(
    const float* __restrict__ emb,
    const int* __restrict__ items,
    const int* __restrict__ user_h0, const int* __restrict__ item_h0,
    const float* __restrict__ eatt,
    float* __restrict__ out, int Btot)
{
    __shared__ float red[64];
    int b = blockIdx.x, d = threadIdx.x;
    float mu = 0.f, mi = 0.f;
    for (int t = 0; t < TT; ++t) {
        mu += emb[(size_t)user_h0[(size_t)b * TT + t] * DIM + d];
        mi += emb[(size_t)item_h0[(size_t)b * TT + t] * DIM + d];
    }
    mu *= (1.f / 64.f); mi *= (1.f / 64.f);
    size_t e0 = (size_t)b * DIM + d;
    size_t cs = (size_t)Btot * DIM;
    float eu = mu + eatt[e0] + eatt[cs + e0];
    float ev = emb[(size_t)items[b] * DIM + d] + eatt[2 * cs + e0] + eatt[3 * cs + e0] + mi;
    red[d] = eu * ev;
    __syncthreads();
    if (d == 0) {
        float s = 0.f;
        for (int i = 0; i < 64; ++i) s += red[i];
        out[b] = 1.f / (1.f + __expf(-s));
    }
}

extern "C" void kernel_launch(void* const* d_in, const int* in_sizes, int n_in,
                              void* d_out, int out_size, void* d_ws, size_t ws_size,
                              hipStream_t stream)
{
    const float* emb    = (const float*)d_in[0];
    const float* rel    = (const float*)d_in[1];
    const float* W1     = (const float*)d_in[2];
    const float* W2     = (const float*)d_in[3];
    const float* W3     = (const float*)d_in[4];
    const int*   items  = (const int*)d_in[5];
    const int*   user_h = (const int*)d_in[6];
    const int*   user_r = (const int*)d_in[7];
    const int*   user_t = (const int*)d_in[8];
    const int*   item_h = (const int*)d_in[9];
    const int*   item_r = (const int*)d_in[10];
    const int*   item_t = (const int*)d_in[11];
    float* out = (float*)d_out;
    const int B = in_sizes[5];

    unsigned int* gW1B = (unsigned int*)d_ws;      // 4096 uints
    unsigned int* gW2B = gW1B + 4096;              // 2048 uints
    float*        eatt = (float*)(gW2B + 2048);    // 4 * B * 64 floats

    ckan_pack_weights<<<12, 256, 0, stream>>>(W1, W2, gW1B, gW2B);
    ckan_attention<<<dim3(B, 4), 128, 0, stream>>>(emb, rel, W3, gW1B, gW2B,
        user_h, user_r, user_t, item_h, item_r, item_t, eatt, B);
    ckan_finalize<<<B, 64, 0, stream>>>(emb, items, user_h, item_h, eatt, out, B);
}
